// Head_76338748719431
// MI455X (gfx1250) — matrix operand order repaired and checked by reading, not compiled, not run on hardware
//
#include <hip/hip_runtime.h>
#include <hip/hip_bf16.h>

// Problem constants (match reference)
constexpr int E_ = 768;
constexpr int H_ = 64;
constexpr int T_ = 4096;
constexpr int B_ = 8;

typedef _Float16 v16h __attribute__((ext_vector_type(16)));
typedef _Float16 h8   __attribute__((ext_vector_type(8)));
typedef float    v8f  __attribute__((ext_vector_type(8)));
typedef unsigned int v4u __attribute__((ext_vector_type(4)));
typedef int      v4i  __attribute__((ext_vector_type(4)));
typedef int      v8i  __attribute__((ext_vector_type(8)));

union F16Frag { v16h v; h8 h[2]; };

__device__ __forceinline__ v8f wmma_f16(v16h a, v16h b, v8f c) {
    // D = A(16x32 f16) * B(32x16 f16) + C(16x16 f32)
    return __builtin_amdgcn_wmma_f32_16x16x32_f16(
        false, a, false, b, (short)0, c, false, false);
}

__device__ __forceinline__ float redmax16(float v) {
#pragma unroll
    for (int m = 1; m < 16; m <<= 1)
        v = fmaxf(v, __shfl_xor(v, m, 32));
    return v;
}
__device__ __forceinline__ float redsum16(float v) {
#pragma unroll
    for (int m = 1; m < 16; m <<= 1)
        v += __shfl_xor(v, m, 32);
    return v;
}

// ---------------------------------------------------------------------------
// Tensor Data Mover: 2D tile load, 16-bit elements.
// D# group0: count=1 | lds_addr | global_addr[56:0] | type=2
// D# group1: data_size=2B, tensor_dim0/1, tile_dim0/1, tensor_dim0_stride
// ---------------------------------------------------------------------------
__device__ __forceinline__ void tdm_load_2d_f16(
    unsigned lds_off, const void* gptr,
    unsigned tile_d0, unsigned tile_d1,
    unsigned tensor_d0, unsigned tensor_d1,
    unsigned long long stride0_elems)
{
    unsigned long long ga = (unsigned long long)gptr;
    v4u g0;
    g0.x = 1u;                                                  // count=1 (valid user D#)
    g0.y = lds_off;                                             // lds_addr (bytes)
    g0.z = (unsigned)ga;                                        // global_addr[31:0]
    g0.w = (unsigned)((ga >> 32) & 0x01FFFFFFu) | (2u << 30);   // addr[56:32] | type=2
    v8i g1;
    g1[0] = (int)(1u << 16);                                    // data_size = 1 -> 2 bytes
    g1[1] = (int)((tensor_d0 & 0xFFFFu) << 16);                 // tensor_dim0[15:0]
    g1[2] = (int)((tensor_d0 >> 16) | ((tensor_d1 & 0xFFFFu) << 16));
    g1[3] = (int)((tensor_d1 >> 16) | ((tile_d0 & 0xFFFFu) << 16));
    g1[4] = (int)(tile_d1 & 0xFFFFu);                           // tile_dim1; tile_dim2=0
    g1[5] = (int)(unsigned)(stride0_elems & 0xFFFFFFFFull);     // dim0 stride[31:0]
    g1[6] = (int)(unsigned)((stride0_elems >> 32) & 0xFFFFull); // dim0 stride[47:32]
    g1[7] = 0;                                                  // dim1 stride unused (2D tile)
    v4i z4 = {};
#if defined(__clang_major__) && (__clang_major__ >= 23)
    v8i z8 = {};
    __builtin_amdgcn_tensor_load_to_lds(g0, g1, z4, z4, z8, 0);
#else
    __builtin_amdgcn_tensor_load_to_lds(g0, g1, z4, z4, 0);
#endif
}

// ---------------------------------------------------------------------------
// Kernel 1: QKV projection.  X(BT x 768) * W(768 x 64) -> Q,K (f16 [g][h]),
// V transposed (f16 [b][h][t]).  64 rows per block, 4 waves x 16 rows.
// ---------------------------------------------------------------------------
__global__ __launch_bounds__(128) void qkv_proj_kernel(
    const float* __restrict__ X,
    const float* __restrict__ Wq,
    const float* __restrict__ Wk,
    const float* __restrict__ Wv,
    _Float16* __restrict__ Qh,
    _Float16* __restrict__ Kh,
    _Float16* __restrict__ Vt)
{
    __shared__ _Float16 Xs[64][40];      // 64 rows x 32 K (pad 8, 16B-aligned runs)
    __shared__ _Float16 Ws[3][64][40];   // transposed: [mat][n][k]

    const int t    = threadIdx.x;
    const int wv   = t >> 5;
    const int lane = t & 31;
    const int lo   = lane & 15;
    const int hi   = lane >> 4;
    const int row0 = blockIdx.x * 64;

    const float* const Wm[3] = {Wq, Wk, Wv};

    v8f acc[3][4] = {};

    for (int kb = 0; kb < E_; kb += 32) {
        if (kb + 32 < E_)   // prefetch next X chunk (one row per thread pair)
            __builtin_prefetch(&X[(size_t)(row0 + (t & 63)) * E_ + kb + 32], 0, 0);

        // Stage X tile (64x32) as f16
#pragma unroll
        for (int i = 0; i < 16; ++i) {
            int idx = t + i * 128;
            int r = idx >> 5, c = idx & 31;
            Xs[r][c] = (_Float16)X[(size_t)(row0 + r) * E_ + kb + c];
        }
        // Stage W tiles transposed (32x64 -> [n][k]) as f16
#pragma unroll
        for (int m = 0; m < 3; ++m) {
#pragma unroll
            for (int i = 0; i < 16; ++i) {
                int idx = t + i * 128;
                int k = idx >> 6, n = idx & 63;
                Ws[m][n][k] = (_Float16)Wm[m][(size_t)(kb + k) * H_ + n];
            }
        }
        __syncthreads();

        // A fragment: rows wv*16 + lo, K = {0..7,16..23} (lanes 0-15) /
        //                                {8..15,24..31} (lanes 16-31)
        F16Frag A;
        {
            const _Float16* xr = &Xs[wv * 16 + lo][0];
            A.h[0] = *(const h8*)(xr + hi * 8);
            A.h[1] = *(const h8*)(xr + 16 + hi * 8);
        }
#pragma unroll
        for (int m = 0; m < 3; ++m) {
#pragma unroll
            for (int nc = 0; nc < 4; ++nc) {
                // B fragment: col = nc*16+lo, K = {hi*8..+7, 16+hi*8..+7}
                F16Frag Bf;
                const _Float16* wr = &Ws[m][nc * 16 + lo][0];
                Bf.h[0] = *(const h8*)(wr + hi * 8);
                Bf.h[1] = *(const h8*)(wr + 16 + hi * 8);
                acc[m][nc] = wmma_f16(A.v, Bf.v, acc[m][nc]);
            }
        }
        __syncthreads();
    }

    // Store: D-frag element r -> row (r + hi*8), col (nc*16 + lo)
#pragma unroll
    for (int nc = 0; nc < 4; ++nc) {
        const int col = nc * 16 + lo;
#pragma unroll
        for (int r = 0; r < 8; ++r) {
            const int g = row0 + wv * 16 + hi * 8 + r;   // global row in [0, B*T)
            Qh[(size_t)g * H_ + col] = (_Float16)acc[0][nc][r];
            Kh[(size_t)g * H_ + col] = (_Float16)acc[1][nc][r];
            const int bidx = g >> 12;                    // T = 4096
            const int tt   = g & (T_ - 1);
            Vt[((size_t)bidx * H_ + col) * T_ + tt] = (_Float16)acc[2][nc][r];
        }
    }
}

// ---------------------------------------------------------------------------
// Kernel 2: causal flash attention.
// Block = 4 waves = 4 consecutive 16-query tiles of one batch.
// K/V 32-key chunks staged in double-buffered LDS by the Tensor Data Mover
// (wave 0 issues descriptors, TENSORcnt + barriers synchronize), shared by
// all 4 waves.  All matmuls on WMMA; online softmax in f32 VALU.
// ---------------------------------------------------------------------------
__global__ __launch_bounds__(128) void attn_kernel(
    const _Float16* __restrict__ Qh,
    const _Float16* __restrict__ Kh,
    const _Float16* __restrict__ Vt,
    float* __restrict__ O)
{
    __shared__ _Float16 Ks[2][32][64];   // [buf][key][h]  (TDM tile, 4KB each)
    __shared__ _Float16 Vs[2][64][32];   // [buf][h][key]  (TDM tile, 4KB each)
    __shared__ _Float16 Pl[4][16][48];   // per-wave P relay (C/D -> A layout)

    const int t    = threadIdx.x;
    const int wv   = t >> 5;
    const int lane = t & 31;
    const int lo   = lane & 15;
    const int hi   = lane >> 4;

    const int qbase = blockIdx.x * 4;          // 4 consecutive q-tiles per block
    const int qtile = qbase + wv;              // [0, B*T/16)
    const int bidx  = qtile >> 8;              // 256 tiles per batch, same for block
    const int q0    = (qtile & 255) * 16;

    const int my_nch  = (q0 + 47) >> 5;                         // my causal chunks
    const int nch_max = ((((qbase + 3) & 255) * 16) + 47) >> 5; // block max chunks

    // Q A-fragments for the two 32-wide halves of H
    F16Frag aq[2];
    {
        const _Float16* qrow = Qh + (size_t)(bidx * T_ + q0 + lo) * H_;
#pragma unroll
        for (int hc = 0; hc < 2; ++hc) {
            aq[hc].h[0] = *(const h8*)(qrow + hc * 32 + hi * 16);
            aq[hc].h[1] = *(const h8*)(qrow + hc * 32 + hi * 16 + 8);
        }
    }

    v8f   o[4] = {};
    float mrow[8], lrow[8];
#pragma unroll
    for (int r = 0; r < 8; ++r) { mrow[r] = -3.4e38f; lrow[r] = 0.0f; }

    // Prologue: TDM-load chunk 0 into buffer 0 (wave 0 only; TDM ignores EXEC)
    if (wv == 0) {
        tdm_load_2d_f16((unsigned)(size_t)&Ks[0][0][0],
                        Kh + (size_t)(bidx * T_) * H_,
                        /*tile*/ 64, 32, /*tensor*/ 64, 32, /*stride*/ 64);
        tdm_load_2d_f16((unsigned)(size_t)&Vs[0][0][0],
                        Vt + (size_t)bidx * H_ * T_,
                        /*tile*/ 32, 64, /*tensor*/ 32, 64, /*stride*/ T_);
    }

    for (int kc = 0; kc < nch_max; ++kc) {
        const int k0  = kc * 32;
        const int cur = kc & 1;
        const int nxt = cur ^ 1;

        if (wv == 0) {
            if (kc + 1 < nch_max) {
                const int k1 = k0 + 32;
                tdm_load_2d_f16((unsigned)(size_t)&Ks[nxt][0][0],
                                Kh + (size_t)(bidx * T_ + k1) * H_,
                                64, 32, 64, 32, 64);
                tdm_load_2d_f16((unsigned)(size_t)&Vs[nxt][0][0],
                                Vt + (size_t)bidx * H_ * T_ + k1,
                                32, 64, 32, 64, T_);
                // keep next chunk's 2 descriptors in flight; current chunk done
                __builtin_amdgcn_s_wait_tensorcnt((short)2);
            } else {
                __builtin_amdgcn_s_wait_tensorcnt((short)0);
            }
        }
        __syncthreads();   // current buffer visible to all waves

        if (kc < my_nch) {
            // S = Q * K^T   (two 16-key sub-tiles, K-dim 64 = 2 WMMAs each)
            v8f s[2];
#pragma unroll
            for (int ks = 0; ks < 2; ++ks) {
                const _Float16* kr = &Ks[cur][ks * 16 + lo][0];
                v16h bk0 = *(const v16h*)(kr + hi * 16);        // h 0..31
                v16h bk1 = *(const v16h*)(kr + 32 + hi * 16);   // h 32..63
                v8f acc = {};
                acc = wmma_f16(aq[0].v, bk0, acc);
                acc = wmma_f16(aq[1].v, bk1, acc);
                s[ks] = acc;
            }

            // Branchless causal mask (q = q0 + r + hi*8, key = k0 + ks*16 + lo)
#pragma unroll
            for (int ks = 0; ks < 2; ++ks) {
                const int key = k0 + ks * 16 + lo;
#pragma unroll
                for (int r = 0; r < 8; ++r) {
                    const int q = q0 + r + hi * 8;
                    s[ks][r] = (key > q) ? -3.0e38f : s[ks][r];
                }
            }

            // Online softmax
            float nm[8], sc[8];
#pragma unroll
            for (int r = 0; r < 8; ++r) {
                float v = redmax16(fmaxf(s[0][r], s[1][r]));
                nm[r] = fmaxf(mrow[r], v);
                sc[r] = __expf(mrow[r] - nm[r]);
                mrow[r] = nm[r];
            }
#pragma unroll
            for (int n = 0; n < 4; ++n)
#pragma unroll
                for (int r = 0; r < 8; ++r)
                    o[n][r] *= sc[r];

            float p0[8], p1[8];
#pragma unroll
            for (int r = 0; r < 8; ++r) {
                p0[r] = __expf(s[0][r] - nm[r]);
                p1[r] = __expf(s[1][r] - nm[r]);
                lrow[r] = lrow[r] * sc[r] + redsum16(p0[r] + p1[r]);
            }

            // Relay P through LDS: C/D layout -> A layout (same wave, DS-ordered)
#pragma unroll
            for (int r = 0; r < 8; ++r) {
                const int row = r + hi * 8;
                Pl[wv][row][lo]      = (_Float16)p0[r];
                Pl[wv][row][16 + lo] = (_Float16)p1[r];
            }
            F16Frag ap;
            {
                const _Float16* pr = &Pl[wv][lo][0];
                ap.h[0] = *(const h8*)(pr + hi * 16);
                ap.h[1] = *(const h8*)(pr + hi * 16 + 8);
            }

            // O += P * V  (V tile is [h][key]: contiguous along keys)
#pragma unroll
            for (int n = 0; n < 4; ++n) {
                const _Float16* vr = &Vs[cur][n * 16 + lo][0];
                v16h bv = *(const v16h*)(vr + hi * 16);
                o[n] = wmma_f16(ap.v, bv, o[n]);
            }
        }

        __syncthreads();   // all waves done with buffer before TDM reuses it
    }

    // Normalize and store fp32 output
#pragma unroll
    for (int n = 0; n < 4; ++n) {
        const int col = n * 16 + lo;
#pragma unroll
        for (int r = 0; r < 8; ++r) {
            const int q = q0 + r + hi * 8;
            O[(size_t)(bidx * T_ + q) * H_ + col] = o[n][r] / lrow[r];
        }
    }
}

// ---------------------------------------------------------------------------
extern "C" void kernel_launch(void* const* d_in, const int* in_sizes, int n_in,
                              void* d_out, int out_size, void* d_ws, size_t ws_size,
                              hipStream_t stream)
{
    const float* X  = (const float*)d_in[0];
    const float* Wq = (const float*)d_in[1];
    const float* Wk = (const float*)d_in[2];
    const float* Wv = (const float*)d_in[3];
    float* O = (float*)d_out;

    const size_t QN = (size_t)B_ * T_ * H_;     // 2,097,152 elements
    _Float16* Qh = (_Float16*)d_ws;             // 4 MB
    _Float16* Kh = Qh + QN;                     // 4 MB
    _Float16* Vt = Kh + QN;                     // 4 MB (transposed [b][h][t])

    // Projection: 32768 rows / 64 per block = 512 blocks
    qkv_proj_kernel<<<512, 128, 0, stream>>>(X, Wq, Wk, Wv, Qh, Kh, Vt);
    // Attention: 2048 query tiles / 4 waves per block = 512 blocks
    attn_kernel<<<512, 128, 0, stream>>>(Qh, Kh, Vt, O);
}
